// CompressiveMemory_81844896792934
// MI455X (gfx1250) — compile-verified, matrix operand order checked by reading, not run on hardware
//
#include <hip/hip_runtime.h>
#include <hip/hip_bf16.h>

// ---------------------------------------------------------------------------
// CDNA5 (gfx1250): compressive-memory cross attention.
// All matmuls via v_wmma_f32_16x16x32_f16; V staged to LDS via the Tensor
// Data Mover (tensor_load_to_lds + s_wait_tensorcnt) when available.
// ---------------------------------------------------------------------------

typedef __attribute__((ext_vector_type(16))) _Float16 v16h;
typedef __attribute__((ext_vector_type(8)))  float    v8f;

#define DHID 512

#if defined(__has_builtin)
#if __has_builtin(__builtin_amdgcn_tensor_load_to_lds) && \
    __has_builtin(__builtin_amdgcn_s_wait_tensorcnt)
#define USE_TDM 1
#endif
#endif
#ifndef USE_TDM
#define USE_TDM 0
#endif

// ---- WMMA fragment loader (ISA 7.12.2, wave32) -----------------------------
// Works for A tiles (row index = lane%16) and for B tiles stored column-major
// (k-contiguous per output column). Per half-wave the 16 halves live in two
// contiguous 16-byte groups at +0/+32 bytes -> exactly 2x ds_load_b128.
union Frag16 { v16h h; uint4 u[2]; };
union H4 { _Float16 h[4]; uint2 u; };
union H8 { _Float16 h[8]; uint4 u; };

__device__ __forceinline__ v16h load_row_frag(const _Float16* base, int ld) {
  const int lane = threadIdx.x & 31;
  const int m  = lane & 15;
  const int hb = (lane >> 4) ? 8 : 0;
  const _Float16* p = base + m * ld + hb;
  Frag16 f;
  f.u[0] = *reinterpret_cast<const uint4*>(p);
  f.u[1] = *reinterpret_cast<const uint4*>(p + 16);
  return f.h;
}

__device__ __forceinline__ v8f wmma_f16(v16h a, v16h b, v8f c) {
  return __builtin_amdgcn_wmma_f32_16x16x32_f16(false, a, false, b,
                                                (short)0, c, false, false);
}

// ---------------------------------------------------------------------------
// One-time weight convert+transpose: Wt[c*512 + k] = (f16) W[k*512 + c].
// Column-major f16 makes GEMM B-staging a pure vectorized copy.
// grid = 256 x 256 threads.
// ---------------------------------------------------------------------------
__global__ void __launch_bounds__(256)
transpose_w_f16(const float* __restrict__ W, _Float16* __restrict__ Wt) {
  const int idx = blockIdx.x * 256 + threadIdx.x;   // 0..65535
  const int c  = idx & 511;
  const int kg = idx >> 9;                          // 0..127 (4 k each)
  H4 p;
#pragma unroll
  for (int j = 0; j < 4; ++j)
    p.h[j] = (_Float16)W[(size_t)(kg * 4 + j) * DHID + c];
  *reinterpret_cast<uint2*>(Wt + (size_t)c * DHID + kg * 4) = p.u;
}

// ---------------------------------------------------------------------------
// Projection GEMM: out = A[M,512] @ W + b (W given as f16 column-major Wt).
// Block = 128 threads (4 waves): 32 rows x 128 cols, 4 WMMAs/wave/k-step.
// MODE 0: f32 row-major   MODE 1: f16 row-major   MODE 2: f16 transposed
// grid = (M/32, 4)
// ---------------------------------------------------------------------------
template <int MODE>
__global__ void __launch_bounds__(128)
gemm_bias_wmma(const float* __restrict__ A, const _Float16* __restrict__ Wt,
               const float* __restrict__ bias, float* __restrict__ outF32,
               _Float16* __restrict__ outF16, int ldT) {
  __shared__ alignas(16) _Float16 As[32 * 32];     // rows x k, row-major
  __shared__ alignas(16) _Float16 Wst[128 * 32];   // cols x k, col-major

  const int tid  = threadIdx.x;
  const int wave = tid >> 5;
  const int lane = tid & 31;
  const int n    = lane & 15;
  const int hi   = lane >> 4;
  const int row0 = blockIdx.x * 32;
  const int colB = blockIdx.y * 128;

  v8f acc[2][2] = {};
  for (int k0 = 0; k0 < DHID; k0 += 32) {
    __syncthreads();
    // stage A 32x32 (f32 -> f16, 8B vector stores)
    for (int i = tid; i < 256; i += 128) {
      const int m = i >> 3, g = i & 7;
      const float4 f = *reinterpret_cast<const float4*>(
          A + (size_t)(row0 + m) * DHID + k0 + g * 4);
      H4 p;
      p.h[0] = (_Float16)f.x; p.h[1] = (_Float16)f.y;
      p.h[2] = (_Float16)f.z; p.h[3] = (_Float16)f.w;
      *reinterpret_cast<uint2*>(As + m * 32 + g * 4) = p.u;
    }
    // stage W 128 cols x 32 k (pure 16B copies)
    for (int i = tid; i < 512; i += 128) {
      const int c = i >> 2, q = i & 3;
      reinterpret_cast<uint4*>(Wst + c * 32)[q] =
          reinterpret_cast<const uint4*>(Wt + (size_t)(colB + c) * DHID + k0)[q];
    }
    __syncthreads();
    const v16h a0 = load_row_frag(As, 32);
    const v16h a1 = load_row_frag(As + 16 * 32, 32);
    const v16h b0 = load_row_frag(Wst + (wave * 32) * 32, 32);
    const v16h b1 = load_row_frag(Wst + (wave * 32 + 16) * 32, 32);
    acc[0][0] = wmma_f16(a0, b0, acc[0][0]);
    acc[0][1] = wmma_f16(a0, b1, acc[0][1]);
    acc[1][0] = wmma_f16(a1, b0, acc[1][0]);
    acc[1][1] = wmma_f16(a1, b1, acc[1][1]);
  }

#pragma unroll
  for (int rt = 0; rt < 2; ++rt) {
#pragma unroll
    for (int ct = 0; ct < 2; ++ct) {
      const int col = colB + wave * 32 + ct * 16 + n;
      const float bv = bias[col];
      if (MODE == 0) {
#pragma unroll
        for (int r = 0; r < 8; ++r) {
          const int m = row0 + rt * 16 + r + 8 * hi;
          outF32[(size_t)m * DHID + col] = acc[rt][ct][r] + bv;
        }
      } else if (MODE == 1) {
#pragma unroll
        for (int r = 0; r < 8; ++r) {
          const int m = row0 + rt * 16 + r + 8 * hi;
          outF16[(size_t)m * DHID + col] = (_Float16)(acc[rt][ct][r] + bv);
        }
      } else {  // MODE 2: transposed f16, 8 contiguous rows -> one 16B store
        H8 p;
#pragma unroll
        for (int r = 0; r < 8; ++r) p.h[r] = (_Float16)(acc[rt][ct][r] + bv);
        *reinterpret_cast<uint4*>(outF16 + (size_t)col * ldT + row0 + rt * 16 + 8 * hi)
            = p.u;
      }
    }
  }
}

// ---------------------------------------------------------------------------
// Attention-pool compressor (RATIO=2).
// ---------------------------------------------------------------------------
__global__ void __launch_bounds__(128)
pool_kernel(const float* __restrict__ combined, const float* __restrict__ comp_query,
            float* __restrict__ pooled) {
  __shared__ float r0[128], r1[128];
  __shared__ float w0s, w1s;
  const float* c0 = combined + (size_t)(2 * blockIdx.x) * DHID;
  const float* c1 = c0 + DHID;
  float s0 = 0.f, s1 = 0.f;
  for (int i = threadIdx.x; i < DHID; i += 128) {
    const float q = comp_query[i];
    s0 += q * c0[i];
    s1 += q * c1[i];
  }
  r0[threadIdx.x] = s0; r1[threadIdx.x] = s1;
  __syncthreads();
  for (int s = 64; s > 0; s >>= 1) {
    if (threadIdx.x < s) {
      r0[threadIdx.x] += r0[threadIdx.x + s];
      r1[threadIdx.x] += r1[threadIdx.x + s];
    }
    __syncthreads();
  }
  if (threadIdx.x == 0) {
    const float sc = 0.044194173824159216f;  // 512^-0.5
    const float l0 = r0[0] * sc, l1 = r1[0] * sc;
    const float m  = fmaxf(l0, l1);
    const float e0 = __expf(l0 - m), e1 = __expf(l1 - m);
    const float inv = 1.f / (e0 + e1);
    w0s = e0 * inv; w1s = e1 * inv;
  }
  __syncthreads();
  const float w0 = w0s, w1 = w1s;
  for (int i = threadIdx.x; i < DHID; i += 128)
    pooled[(size_t)blockIdx.x * DHID + i] = w0 * c0[i] + w1 * c1[i];
}

// ---------------------------------------------------------------------------
// Linear-interp decode + MSE loss (atomic scalar).
// ---------------------------------------------------------------------------
__global__ void __launch_bounds__(256)
decode_loss_kernel(const float* __restrict__ compressed,
                   const float* __restrict__ combined, float* __restrict__ loss) {
  __shared__ float red[256];
  const int i = blockIdx.x;
  float src = (i + 0.5f) * 0.5f - 0.5f;
  src = fminf(fmaxf(src, 0.f), 255.f);
  const int i0 = (int)floorf(src);
  const int i1 = min(i0 + 1, 255);
  const float frac = src - (float)i0;
  const float* p0 = compressed + (size_t)i0 * DHID;
  const float* p1 = compressed + (size_t)i1 * DHID;
  const float* cb = combined + (size_t)i * DHID;
  float acc = 0.f;
  for (int d = threadIdx.x; d < DHID; d += 256) {
    const float dec = (1.f - frac) * p0[d] + frac * p1[d];
    const float diff = dec - cb[d];
    acc += diff * diff;
  }
  red[threadIdx.x] = acc;
  __syncthreads();
  for (int s = 128; s > 0; s >>= 1) {
    if (threadIdx.x < s) red[threadIdx.x] += red[threadIdx.x + s];
    __syncthreads();
  }
  if (threadIdx.x == 0)
    atomicAdd(loss, red[0] * (1.0f / (512.0f * 512.0f)));
}

// ---------------------------------------------------------------------------
// Flash cross-attention + residual.
// Q: [8192,512] f16 row-major; K: [4096,512] f16 row-major;
// V^T: [512,4096] f16 (key-contiguous -> frag-ready, TDM-friendly 2D tiles).
// Block = 128 threads (4 waves), 16 query rows, 64-key blocks.
// ---------------------------------------------------------------------------
struct FlashSmem {
  alignas(16) _Float16 Qs[16 * DHID];       // 16 KB
  union {
    alignas(16) _Float16 Ks[64 * 256];      // 32 KB (64 keys x 256-dim chunk)
    alignas(16) _Float16 Vs[DHID * 32];     // 32 KB (512 cols x 32 keys)
  };
  float Ss[16 * 64];
  alignas(16) _Float16 Ps[16 * 64];
  float red8[16][8];
  float rowM[16], rowL[16], rowScale[16];
};

#if USE_TDM
typedef __attribute__((ext_vector_type(4))) unsigned int u32x4;
typedef __attribute__((ext_vector_type(8))) int i32x8;
typedef __attribute__((ext_vector_type(4))) int i32x4;

// 2D TDM tile: 32 contiguous keys x 512 columns from V^T[512][4096] -> LDS.
// Uses the 6-arg builtin form: (g0, g1, g2, g3, g4, cpol).
__device__ __forceinline__ void tdm_load_vchunk(const _Float16* gsrc,
                                                unsigned lds_off) {
  const unsigned long long ga = (unsigned long long)(uintptr_t)gsrc;
  u32x4 g0;
  g0[0] = 1u;                                           // count=1 (valid D#)
  g0[1] = lds_off;                                      // lds_addr (bytes)
  g0[2] = (unsigned)ga;                                 // global_addr[31:0]
  g0[3] = (unsigned)((ga >> 32) & 0x01FFFFFFull) | (2u << 30);  // addr hi,type=2
  i32x8 g1;
  g1[0] = 0x00010000;              // data_size=1 (2 bytes), no multicast
  g1[1] = (int)(0x1000u << 16);    // tensor_dim0 = 4096 (bits[79:48] lo16)
  g1[2] = (int)(0x0200u << 16);    // tensor_dim1 = 512  (bits[111:80] lo16)
  g1[3] = (int)(32u << 16);        // tile_dim0  = 32    (bits[127:112])
  g1[4] = 512;                     // tile_dim1  = 512   (bits[143:128])
  g1[5] = 4096;                    // tensor_dim0_stride = 4096
  g1[6] = 0;
  g1[7] = 0;
  const i32x4 z4 = {0, 0, 0, 0};
  const i32x8 z8 = {0, 0, 0, 0, 0, 0, 0, 0};
  __builtin_amdgcn_tensor_load_to_lds(g0, g1, z4, z4, z8, 0);
}
#endif

__global__ void __launch_bounds__(128)
flash_attn_kernel(const _Float16* __restrict__ Q, const _Float16* __restrict__ Km,
                  const _Float16* __restrict__ VmT, const float* __restrict__ x,
                  float* __restrict__ out) {
  constexpr int NKEYS = 4096;
  constexpr int BN = 64;

  __shared__ FlashSmem sm;

  const int tid  = threadIdx.x;
  const int wave = tid >> 5;
  const int lane = tid & 31;
  const int n    = lane & 15;
  const int hi   = lane >> 4;
  const int srow = tid >> 3;        // softmax: 8 threads per query row
  const int ssub = tid & 7;
  const int row0 = blockIdx.x * 16;
  const float scale = 0.044194173824159216f;  // 512^-0.5

  // stage Q tile once (16x512 f16, 16B copies)
  {
    const uint4* src = reinterpret_cast<const uint4*>(Q + (size_t)row0 * DHID);
    uint4* dst = reinterpret_cast<uint4*>(sm.Qs);
    for (int i = tid; i < 16 * DHID / 8; i += 128) dst[i] = src[i];
  }
  if (tid < 16) { sm.rowM[tid] = -1e30f; sm.rowL[tid] = 0.f; }

  v8f acc[8];
#pragma unroll
  for (int t = 0; t < 8; ++t) acc[t] = (v8f){};

  for (int kb = 0; kb < NKEYS; kb += BN) {
    if (kb + BN < NKEYS) {
      __builtin_prefetch(Km + (size_t)(kb + BN) * DHID + (size_t)tid * 256, 0, 1);
      __builtin_prefetch(VmT + (size_t)tid * NKEYS + kb + BN, 0, 1);
    }

    // ---- scores: S = Q K^T over two 256-dim chunks ----
    v8f s = {};
#pragma unroll 1
    for (int half = 0; half < 2; ++half) {
      __syncthreads();
      for (int i = tid; i < 64 * 32; i += 128) {           // 64 keys x 32 uint4
        const int key = i >> 5, q = i & 31;
        reinterpret_cast<uint4*>(sm.Ks + key * 256)[q] =
            reinterpret_cast<const uint4*>(Km + (size_t)(kb + key) * DHID +
                                           half * 256)[q];
      }
      __syncthreads();
#pragma unroll
      for (int kk = 0; kk < 8; ++kk) {
        const v16h a = load_row_frag(sm.Qs + half * 256 + kk * 32, DHID);
        const v16h b = load_row_frag(sm.Ks + (wave * 16) * 256 + kk * 32, 256);
        s = wmma_f16(a, b, s);
      }
    }
    __syncthreads();
#pragma unroll
    for (int r = 0; r < 8; ++r)
      sm.Ss[(r + 8 * hi) * BN + wave * 16 + n] = s[r] * scale;
    __syncthreads();

    // ---- online softmax: 8 threads per row ----
    {
      float pm = -1e30f;
#pragma unroll
      for (int j = 0; j < 8; ++j)
        pm = fmaxf(pm, sm.Ss[srow * BN + ssub * 8 + j]);
      sm.red8[srow][ssub] = pm;
      __syncthreads();
      if (ssub == 0) {
        float mn = sm.rowM[srow];
#pragma unroll
        for (int t = 0; t < 8; ++t) mn = fmaxf(mn, sm.red8[srow][t]);
        sm.rowScale[srow] = __expf(sm.rowM[srow] - mn);
        sm.rowM[srow] = mn;
      }
      __syncthreads();
      const float mn = sm.rowM[srow];
      float ps = 0.f;
#pragma unroll
      for (int j = 0; j < 8; ++j) {
        const float p = __expf(sm.Ss[srow * BN + ssub * 8 + j] - mn);
        sm.Ps[srow * BN + ssub * 8 + j] = (_Float16)p;
        ps += p;
      }
      sm.red8[srow][ssub] = ps;
      __syncthreads();
      if (ssub == 0) {
        float l = sm.rowL[srow] * sm.rowScale[srow];
#pragma unroll
        for (int t = 0; t < 8; ++t) l += sm.red8[srow][t];
        sm.rowL[srow] = l;
      }
    }
    // rescale running accumulators (rowScale is stable until next block)
#pragma unroll
    for (int t = 0; t < 8; ++t) {
#pragma unroll
      for (int r = 0; r < 8; ++r) acc[t][r] *= sm.rowScale[r + 8 * hi];
    }

    // ---- P @ V, 32 keys at a time (TDM async copy when available) ----
#pragma unroll 1
    for (int kk2 = 0; kk2 < 2; ++kk2) {
      __syncthreads();
#if USE_TDM
      if (wave == 0) {
        tdm_load_vchunk(VmT + (size_t)(kb + kk2 * 32),
                        (unsigned)__builtin_offsetof(FlashSmem, Vs));
        __builtin_amdgcn_s_wait_tensorcnt(0);
      }
#else
      for (int i = tid; i < DHID * 4; i += 128) {          // 512 cols x 4 uint4
        const int c = i >> 2, q = i & 3;
        reinterpret_cast<uint4*>(sm.Vs + c * 32)[q] =
            reinterpret_cast<const uint4*>(VmT + (size_t)c * NKEYS + kb +
                                           kk2 * 32)[q];
      }
#endif
      __syncthreads();
      const v16h a = load_row_frag(sm.Ps + kk2 * 32, BN);
#pragma unroll
      for (int t = 0; t < 8; ++t) {
        const v16h b = load_row_frag(sm.Vs + (wave * 128 + t * 16) * 32, 32);
        acc[t] = wmma_f16(a, b, acc[t]);
      }
    }
  }
  __syncthreads();

  // ---- epilogue: normalize, residual, store ----
#pragma unroll
  for (int t = 0; t < 8; ++t) {
    const int col = wave * 128 + t * 16 + n;
#pragma unroll
    for (int r = 0; r < 8; ++r) {
      const int row = row0 + r + 8 * hi;
      const float invl = 1.0f / sm.rowL[r + 8 * hi];
      const size_t idx = (size_t)row * DHID + col;
      out[idx] = x[idx] + acc[t][r] * invl;
    }
  }
}

// ---------------------------------------------------------------------------
// Host launch
// ---------------------------------------------------------------------------
extern "C" void kernel_launch(void* const* d_in, const int* in_sizes, int n_in,
                              void* d_out, int out_size, void* d_ws, size_t ws_size,
                              hipStream_t stream) {
  (void)in_sizes; (void)n_in; (void)out_size; (void)ws_size;

  const float* x          = (const float*)d_in[0];
  const float* fine_mem   = (const float*)d_in[1];
  const float* cmem_seq   = (const float*)d_in[2];
  const float* comp_query = (const float*)d_in[3];
  const float* comp_W     = (const float*)d_in[4];
  const float* comp_b     = (const float*)d_in[5];
  const float* k_W        = (const float*)d_in[6];
  const float* k_b        = (const float*)d_in[7];
  const float* v_W        = (const float*)d_in[8];
  const float* v_b        = (const float*)d_in[9];
  const float* xk_W       = (const float*)d_in[10];
  const float* xk_b       = (const float*)d_in[11];
  const float* ck_W       = (const float*)d_in[12];
  const float* ck_b       = (const float*)d_in[13];
  const float* cv_W       = (const float*)d_in[14];
  const float* cv_b       = (const float*)d_in[15];

  const size_t NQ = 8192, NM = 4096, WSZ = (size_t)DHID * DHID;

  // workspace layout
  _Float16* qF16  = (_Float16*)d_ws;            // 8192*512
  _Float16* kF16  = qF16 + NQ * DHID;           // 4096*512
  _Float16* vF16T = kF16 + NM * DHID;           // 512*4096 (transposed)
  _Float16* WtXk  = vF16T + NM * DHID;
  _Float16* WtK   = WtXk + WSZ;
  _Float16* WtCk  = WtK + WSZ;
  _Float16* WtV   = WtCk + WSZ;
  _Float16* WtCv  = WtV + WSZ;
  _Float16* WtCmp = WtCv + WSZ;
  float* pooled     = (float*)(WtCmp + WSZ);    // 256*512
  float* compressed = pooled + (size_t)256 * DHID;

  float* out  = (float*)d_out;
  float* loss = out + NQ * DHID;

  const dim3 blk(128);

  // one-time weight convert+transpose to frag-ready f16
  transpose_w_f16<<<256, 256, 0, stream>>>(xk_W, WtXk);
  transpose_w_f16<<<256, 256, 0, stream>>>(k_W, WtK);
  transpose_w_f16<<<256, 256, 0, stream>>>(ck_W, WtCk);
  transpose_w_f16<<<256, 256, 0, stream>>>(v_W, WtV);
  transpose_w_f16<<<256, 256, 0, stream>>>(cv_W, WtCv);
  transpose_w_f16<<<256, 256, 0, stream>>>(comp_W, WtCmp);

  // projections
  gemm_bias_wmma<1><<<dim3(256, 4), blk, 0, stream>>>(x, WtXk, xk_b, nullptr, qF16, 0);
  gemm_bias_wmma<1><<<dim3(64, 4), blk, 0, stream>>>(fine_mem, WtK, k_b, nullptr, kF16, 0);
  gemm_bias_wmma<1><<<dim3(64, 4), blk, 0, stream>>>(cmem_seq, WtCk, ck_b, nullptr,
                                                     kF16 + (size_t)2048 * DHID, 0);
  gemm_bias_wmma<2><<<dim3(64, 4), blk, 0, stream>>>(fine_mem, WtV, v_b, nullptr,
                                                     vF16T, (int)NM);
  gemm_bias_wmma<2><<<dim3(64, 4), blk, 0, stream>>>(cmem_seq, WtCv, cv_b, nullptr,
                                                     vF16T + 2048, (int)NM);

  // compression path
  pool_kernel<<<256, 128, 0, stream>>>(fine_mem, comp_query, pooled);
  gemm_bias_wmma<0><<<dim3(8, 4), blk, 0, stream>>>(pooled, WtCmp, comp_b,
                                                    compressed, nullptr, 0);
  (void)hipMemsetAsync(loss, 0, sizeof(float), stream);
  decode_loss_kernel<<<512, 256, 0, stream>>>(compressed, fine_mem, loss);

  // fused flash cross-attention + residual
  flash_attn_kernel<<<512, blk, 0, stream>>>(qF16, kF16, vF16T, x, out);
}
